// FILR_20418274526024
// MI455X (gfx1250) — matrix operand order repaired and checked by reading, not compiled, not run on hardware
//
#include <hip/hip_runtime.h>
#include <math.h>

typedef __attribute__((ext_vector_type(16))) __bf16 v16bf;
typedef __attribute__((ext_vector_type(8)))  __bf16 v8bf;
typedef __attribute__((ext_vector_type(8)))  float  v8f;

#define EPSF    1e-12f
#define LN2PIF  1.8378770664093453f

// =====================================================================
// Layout prep: NCHW f32 -> NHWC bf16  (B fragments become contiguous)
// =====================================================================
__global__ void nchw_to_nhwc_bf16(const float* __restrict__ src,
    __bf16* __restrict__ dst, int C, int H, int W, int total)
{
    int i = blockIdx.x * 256 + threadIdx.x;
    if (i >= total) return;
    int c = i % C, x = (i / C) % W, y = (i / (C * W)) % H, b = i / (C * W * H);
    dst[i] = (__bf16)src[(((size_t)b * C + c) * H + y) * W + x];
}

// OIHW f32 -> bf16 with K permuted as k' = (kh*KW+kw)*Cin + ci
__global__ void wgt_to_bf16(const float* __restrict__ w, __bf16* __restrict__ dst,
    int Cin, int KK, int total)
{
    int i = blockIdx.x * 256 + threadIdx.x;
    if (i >= total) return;
    int ci = i % Cin, cell = (i / Cin) % KK, m = i / (Cin * KK);
    dst[i] = (__bf16)w[((size_t)m * Cin + ci) * KK + cell];
}

// =====================================================================
// WMMA implicit-GEMM convolution (stride 1).
// Requires Cin % 32 == 0 (holds for every conv in this net).
// GEMM: M = Cout, N = B*Hout*Wout, K = Cin*KH*KW (cell-major, ci inner).
// 4 waves / block, each wave owns one 16x16 output tile.
// Per 32-wide K chunk: 2 b128 loads (A) + 2 b128 loads (B) + 1 WMMA.
// =====================================================================
__global__ void __launch_bounds__(128) wmma_conv(
    const __bf16* __restrict__ nhwc, const __bf16* __restrict__ wgtT,
    const float* __restrict__ bias, float* __restrict__ out,
    int Bn, int Cin, int H, int W, int Cout, int KH, int KW,
    int pad, int Hout, int Wout)
{
    const int lane = threadIdx.x & 31;
    const int widx = threadIdx.x >> 5;
    const int g    = lane >> 4;      // 16-lane half
    const int hl   = lane & 15;

    const int N    = Bn * Hout * Wout;
    const int KK   = KH * KW;
    const int Ktot = Cin * KK;

    // output column (pixel): clamp for loads, mask at store
    int n = (blockIdx.x * 4 + widx) * 16 + hl;
    const bool valid_n = (n < N);
    int n_c = valid_n ? n : (N - 1);
    int b   = n_c / (Hout * Wout);
    int rem = n_c % (Hout * Wout);
    int oy  = rem / Wout, ox = rem % Wout;

    // A row (output channel): clamp for loads, mask at store
    int mA   = blockIdx.y * 16 + hl;
    int mA_c = (mA < Cout) ? mA : (Cout - 1);
    const __bf16* aprow = wgtT + (size_t)mA_c * Ktot;

    v8f acc = {};
    const v8bf vz = {};

    for (int cell = 0; cell < KK; ++cell) {
        int kh = cell / KW, kw = cell % KW;
        int y = oy + kh - pad, x = ox + kw - pad;
        bool ok = valid_n && (y >= 0) && (y < H) && (x >= 0) && (x < W);
        int y_c = y < 0 ? 0 : (y >= H ? H - 1 : y);
        int x_c = x < 0 ? 0 : (x >= W ? W - 1 : x);
        const __bf16* bp = nhwc + (((size_t)b * H + y_c) * W + x_c) * Cin;
        const __bf16* ap = aprow + (size_t)cell * Cin;
        if (cell + 1 < KK)
            __builtin_prefetch(ap + Cin, 0, 3);

        for (int ci = 0; ci < Cin; ci += 32) {
            // A fragment: two contiguous 8-bf16 runs (16B b128 loads)
            v8bf a1 = *(const v8bf*)(ap + ci + g * 8);
            v8bf a2 = *(const v8bf*)(ap + ci + 16 + g * 8);
            // B fragment: two contiguous 8-bf16 runs (16B b128 loads)
            v8bf b1 = *(const v8bf*)(bp + ci + g * 8);
            v8bf b2 = *(const v8bf*)(bp + ci + 16 + g * 8);
            b1 = ok ? b1 : vz;
            b2 = ok ? b2 : vz;
            v16bf af = __builtin_shufflevector(a1, a2,
                0, 1, 2, 3, 4, 5, 6, 7, 8, 9, 10, 11, 12, 13, 14, 15);
            v16bf bfv = __builtin_shufflevector(b1, b2,
                0, 1, 2, 3, 4, 5, 6, 7, 8, 9, 10, 11, 12, 13, 14, 15);
            acc = __builtin_amdgcn_wmma_f32_16x16x32_bf16(
                false, af, false, bfv, (short)0, acc, false, false);
        }
    }

    // D layout: acc[v] is (M = tileM + v + g*8, N = n)
    #pragma unroll
    for (int v = 0; v < 8; ++v) {
        int m = blockIdx.y * 16 + v + g * 8;
        if (m < Cout && valid_n) {
            float val = acc[v] + (bias ? bias[m] : 0.f);
            out[(((size_t)b * Cout + m) * Hout + oy) * Wout + ox] = val;
        }
    }
}

// =====================================================================
// BatchNorm (training-mode batch statistics over (B,H,W) per channel)
// =====================================================================
__global__ void __launch_bounds__(256) bn_stats(
    const float* __restrict__ x, float* __restrict__ mean,
    float* __restrict__ var, int Bn, int C, int HW)
{
    int c = blockIdx.x, tid = threadIdx.x;
    float s = 0.f, s2 = 0.f;
    for (int b = 0; b < Bn; ++b) {
        const float* p = x + ((size_t)b * C + c) * HW;
        for (int i = tid; i < HW; i += 256) { float v = p[i]; s += v; s2 += v * v; }
    }
    __shared__ float sh[256], sh2[256];
    sh[tid] = s; sh2[tid] = s2; __syncthreads();
    for (int st = 128; st > 0; st >>= 1) {
        if (tid < st) { sh[tid] += sh[tid + st]; sh2[tid] += sh2[tid + st]; }
        __syncthreads();
    }
    if (tid == 0) {
        float n = (float)(Bn * HW);
        float m = sh[0] / n;
        mean[c] = m; var[c] = sh2[0] / n - m * m;
    }
}

// act: 0 = none, 1 = relu, 2 = sigmoid
__global__ void bn_apply(float* __restrict__ x,
    const float* __restrict__ mean, const float* __restrict__ var,
    const float* __restrict__ g, const float* __restrict__ bt,
    int C, int HW, int total, int act)
{
    int i = blockIdx.x * 256 + threadIdx.x;
    if (i >= total) return;
    int c = (i / HW) % C;
    float y = (x[i] - mean[c]) * rsqrtf(var[c] + 1e-5f) * g[c] + bt[c];
    if (act == 1) y = fmaxf(y, 0.f);
    else if (act == 2) y = 1.f / (1.f + expf(-y));
    x[i] = y;
}

// =====================================================================
// Bilinear resize (align-corners, matches _interp_mat) into a channel
// window [c0, c0+C) of a (Bn, Cdst, ho, wo) destination.
// =====================================================================
__global__ void resize_into(const float* __restrict__ src, float* __restrict__ dst,
    int Bn, int C, int hi, int wi, int ho, int wo, int Cdst, int c0)
{
    int i = blockIdx.x * 256 + threadIdx.x;
    int total = Bn * C * ho * wo;
    if (i >= total) return;
    int x = i % wo, y = (i / wo) % ho;
    int c = (i / (wo * ho)) % C, b = i / (wo * ho * C);
    float sy = (ho > 1) ? (float)(hi - 1) / (float)(ho - 1) : 0.f;
    float sx = (wo > 1) ? (float)(wi - 1) / (float)(wo - 1) : 0.f;
    float cy = y * sy, cx = x * sx;
    int y0 = (int)floorf(cy); y0 = y0 < 0 ? 0 : (y0 > hi - 1 ? hi - 1 : y0);
    int x0 = (int)floorf(cx); x0 = x0 < 0 ? 0 : (x0 > wi - 1 ? wi - 1 : x0);
    int y1 = y0 + 1 < hi ? y0 + 1 : hi - 1;
    int x1 = x0 + 1 < wi ? x0 + 1 : wi - 1;
    float ty = cy - (float)y0, tx = cx - (float)x0;
    const float* p = src + ((size_t)b * C + c) * hi * wi;
    float v00 = p[y0 * wi + x0], v01 = p[y0 * wi + x1];
    float v10 = p[y1 * wi + x0], v11 = p[y1 * wi + x1];
    float v = (1.f - ty) * ((1.f - tx) * v00 + tx * v01)
            +        ty  * ((1.f - tx) * v10 + tx * v11);
    dst[(((size_t)b * Cdst + c0 + c) * ho + y) * wo + x] = v;
}

__global__ void copy_into(const float* __restrict__ src, float* __restrict__ dst,
    int Bn, int C, int HW, int Cdst, int c0)
{
    int i = blockIdx.x * 256 + threadIdx.x;
    int total = Bn * C * HW;
    if (i >= total) return;
    int s = i % HW, c = (i / HW) % C, b = i / (HW * C);
    dst[((size_t)b * Cdst + c0 + c) * HW + s] = src[i];
}

// =====================================================================
// Channel attention helpers
// =====================================================================
__global__ void __launch_bounds__(256) chan_spatial_max(
    const float* __restrict__ x, float* __restrict__ mp, int HW)
{
    int bc = blockIdx.x, tid = threadIdx.x;
    const float* p = x + (size_t)bc * HW;
    float m = -3.4e38f;
    for (int i = tid; i < HW; i += 256) m = fmaxf(m, p[i]);
    __shared__ float sh[256];
    sh[tid] = m; __syncthreads();
    for (int st = 128; st > 0; st >>= 1) {
        if (tid < st) sh[tid] = fmaxf(sh[tid], sh[tid + st]);
        __syncthreads();
    }
    if (tid == 0) mp[bc] = sh[0];
}

__global__ void __launch_bounds__(192) ca_kernel(
    const float* __restrict__ mp, const float* __restrict__ fc1,
    const float* __restrict__ fc2, float* __restrict__ ca)
{
    int b = blockIdx.x, t = threadIdx.x;
    __shared__ float hid[12];
    if (t < 12) {
        float s = 0.f;
        for (int c = 0; c < 192; ++c) s += mp[b * 192 + c] * fc1[t * 192 + c];
        hid[t] = fmaxf(s, 0.f);
    }
    __syncthreads();
    float s = 0.f;
    for (int j = 0; j < 12; ++j) s += hid[j] * fc2[t * 12 + j];
    ca[b * 192 + t] = 1.f / (1.f + expf(-s));
}

__global__ void scale_ca(float* __restrict__ x, const float* __restrict__ ca,
                         int HW, int total)
{
    int i = blockIdx.x * 256 + threadIdx.x;
    if (i >= total) return;
    int bc = i / HW;
    x[i] += x[i] * ca[bc];
}

// =====================================================================
// Spatial attention: max over channels, 7x7 single-channel conv+sigmoid,
// x = x + x*sa
// =====================================================================
__global__ void max_over_c(const float* __restrict__ x, float* __restrict__ o,
                           int C, int HW, int total)
{
    int i = blockIdx.x * 256 + threadIdx.x;
    if (i >= total) return;
    int b = i / HW, s = i % HW;
    float m = -3.4e38f;
    for (int c = 0; c < C; ++c) m = fmaxf(m, x[((size_t)b * C + c) * HW + s]);
    o[i] = m;
}

__global__ void conv7_sig(const float* __restrict__ x, const float* __restrict__ w,
                          float* __restrict__ o, int H, int W, int total)
{
    int i = blockIdx.x * 256 + threadIdx.x;
    if (i >= total) return;
    int b = i / (H * W), y = (i / W) % H, xx = i % W;
    float s = 0.f;
    for (int kh = 0; kh < 7; ++kh)
        for (int kw = 0; kw < 7; ++kw) {
            int yy = y + kh - 3, xw = xx + kw - 3;
            if (yy >= 0 && yy < H && xw >= 0 && xw < W)
                s += x[(size_t)b * H * W + yy * W + xw] * w[kh * 7 + kw];
        }
    o[i] = 1.f / (1.f + expf(-s));
}

__global__ void apply_sa(float* __restrict__ x, const float* __restrict__ sa,
                         int C, int HW, int total)
{
    int i = blockIdx.x * 256 + threadIdx.x;
    if (i >= total) return;
    int b = i / (C * HW), s = i % HW;
    float a = sa[b * HW + s];
    x[i] += x[i] * a;
}

__global__ void ew_mul(const float* __restrict__ a, const float* __restrict__ b,
                       float* __restrict__ o, int total)
{
    int i = blockIdx.x * 256 + threadIdx.x;
    if (i < total) o[i] = a[i] * b[i];
}

__global__ void ew_add_into(float* __restrict__ a, const float* __restrict__ b, int total)
{
    int i = blockIdx.x * 256 + threadIdx.x;
    if (i < total) a[i] += b[i];
}

// =====================================================================
// EM routing: one workgroup per (batch, output position).
// A=8, B=8, psize=16, k=3, stride=2, pad=1, iters=3
// input m: (4,8,46,46)  pose: (4,128,46,46)  ->  pose_out: (4,128,23,23)
// Entire working set (~47 KB) lives in LDS.
// =====================================================================
__global__ void __launch_bounds__(128) em_routing_kernel(
    const float* __restrict__ m, const float* __restrict__ pose,
    const float* __restrict__ emW, const float* __restrict__ beta_u,
    const float* __restrict__ beta_a, float* __restrict__ pose_out)
{
    const int L = 23, Hin = 46;
    int blk = blockIdx.x;
    int b = blk / (L * L), l = blk % (L * L);
    int oy = l / L, ox = l % L;
    int tid = threadIdx.x;

    __shared__ float pu[72][16];
    __shared__ float vsh[72][8][16];
    __shared__ float au[72];
    __shared__ float rr[72][8];
    __shared__ float mu[8][16];
    __shared__ float sig[8][16];
    __shared__ float rsum[8];
    __shared__ float aout[8];
    __shared__ float lnp[72][8];

    for (int idx = tid; idx < 72 * 16; idx += 128) {
        int j = idx >> 4, p = idx & 15;
        int kk = j >> 3, a = j & 7;
        int kh = kk / 3, kw = kk % 3;
        int y = oy * 2 - 1 + kh, x = ox * 2 - 1 + kw;
        float val = 0.f;
        if (y >= 0 && y < Hin && x >= 0 && x < Hin)
            val = pose[(((size_t)b * 128 + (a * 16 + p)) * Hin + y) * Hin + x];
        pu[j][p] = val;
    }
    for (int j = tid; j < 72; j += 128) {
        int kk = j >> 3, a = j & 7;
        int kh = kk / 3, kw = kk % 3;
        int y = oy * 2 - 1 + kh, x = ox * 2 - 1 + kw;
        float val = 0.f;
        if (y >= 0 && y < Hin && x >= 0 && x < Hin)
            val = m[(((size_t)b * 8 + a) * Hin + y) * Hin + x];
        au[j] = val;
        for (int o = 0; o < 8; ++o) rr[j][o] = 1.f;
    }
    __syncthreads();

    for (int idx = tid; idx < 72 * 8 * 16; idx += 128) {
        int j = idx >> 7, o = (idx >> 4) & 7, p = idx & 15;
        int row = p >> 2, col = p & 3;
        float s = 0.f;
        for (int t = 0; t < 4; ++t)
            s += pu[j][row * 4 + t] * emW[(((size_t)j * 8 + o) * 4 + t) * 4 + col];
        vsh[j][o][p] = s;
    }
    __syncthreads();

    for (int it = 0; it < 3; ++it) {
        float lam = 0.01f * (1.f - powf(0.95f, (float)(it + 1)));
        if (tid < 8) {
            float s = 0.f;
            for (int j = 0; j < 72; ++j) s += rr[j][tid] * au[j];
            rsum[tid] = s;
        }
        __syncthreads();
        {
            int o = tid >> 4, p = tid & 15;   // 128 threads == 8*16
            float s = 0.f;
            for (int j = 0; j < 72; ++j)
                s += (rr[j][o] * au[j]) / (rsum[o] + EPSF) * vsh[j][o][p];
            mu[o][p] = s;
        }
        __syncthreads();
        {
            int o = tid >> 4, p = tid & 15;
            float s = 0.f;
            for (int j = 0; j < 72; ++j) {
                float d = vsh[j][o][p] - mu[o][p];
                s += (rr[j][o] * au[j]) / (rsum[o] + EPSF) * d * d;
            }
            sig[o][p] = s + EPSF;
        }
        __syncthreads();
        if (tid < 8) {
            int o = tid;
            float c = 0.f;
            for (int p = 0; p < 16; ++p)
                c += (beta_u[o] + 0.5f * logf(sig[o][p])) * rsum[o];
            aout[o] = 1.f / (1.f + expf(-lam * (beta_a[o] - c)));
        }
        __syncthreads();
        if (it < 2) {
            for (int idx = tid; idx < 72 * 8; idx += 128) {
                int j = idx >> 3, o = idx & 7;
                float s1 = 0.f, s2 = 0.f;
                for (int p = 0; p < 16; ++p) {
                    s1 += logf(sig[o][p] * LN2PIF);
                    float d = vsh[j][o][p] - mu[o][p];
                    s2 += d * d / (2.f * sig[o][p]);
                }
                lnp[j][o] = -0.5f * s1 - s2 + logf(aout[o]);
            }
            __syncthreads();
            for (int j = tid; j < 72; j += 128) {
                float mx = -3.4e38f;
                for (int o = 0; o < 8; ++o) mx = fmaxf(mx, lnp[j][o]);
                float den = 0.f;
                for (int o = 0; o < 8; ++o) {
                    float e = expf(lnp[j][o] - mx);
                    rr[j][o] = e; den += e;
                }
                for (int o = 0; o < 8; ++o) rr[j][o] /= den;
            }
            __syncthreads();
        }
    }
    {
        int o = tid >> 4, p = tid & 15;
        pose_out[(((size_t)b * 128 + (o * 16 + p)) * L + oy) * L + ox] = mu[o][p];
    }
}

// =====================================================================
// Host orchestration
// =====================================================================
#define EWG(n) ((n) + 255) / 256, 256, 0, s

static inline void launch_conv(hipStream_t s, const float* in, const float* w,
    const float* b, float* out, __bf16* nhwc, __bf16* wgtT,
    int Bn, int Cin, int H, int W, int Cout, int KH, int KW, int pad)
{
    int tin = Bn * Cin * H * W;
    nchw_to_nhwc_bf16<<<EWG(tin)>>>(in, nhwc, Cin, H, W, tin);
    int KK = KH * KW;
    int tw = Cout * Cin * KK;
    wgt_to_bf16<<<EWG(tw)>>>(w, wgtT, Cin, KK, tw);

    int Hout = H + 2 * pad - KH + 1;
    int Wout = W + 2 * pad - KW + 1;
    int N = Bn * Hout * Wout;
    int Ntiles = (N + 15) / 16;
    dim3 grid((Ntiles + 3) / 4, (Cout + 15) / 16);
    wmma_conv<<<grid, 128, 0, s>>>(nhwc, wgtT, b, out, Bn, Cin, H, W,
                                   Cout, KH, KW, pad, Hout, Wout);
}

static inline void run_bn(hipStream_t s, float* x, float* mean, float* var,
    const float* g, const float* bt, int Bn, int C, int HW, int act)
{
    bn_stats<<<C, 256, 0, s>>>(x, mean, var, Bn, C, HW);
    int total = Bn * C * HW;
    bn_apply<<<(total + 255) / 256, 256, 0, s>>>(x, mean, var, g, bt, C, HW, total, act);
}

extern "C" void kernel_launch(void* const* d_in, const int* in_sizes, int n_in,
                              void* d_out, int out_size, void* d_ws, size_t ws_size,
                              hipStream_t stream)
{
    (void)in_sizes; (void)n_in; (void)out_size; (void)ws_size;
    hipStream_t s = stream;

    const float* in1 = (const float*)d_in[0];   // (4,64,24,24)
    const float* in2 = (const float*)d_in[1];   // (4,64,48,48)
    const float* in3 = (const float*)d_in[2];   // (4,64,96,96)
    int pi = 3;
    const float* ca_fc1  = (const float*)d_in[pi++];
    const float* ca_fc2  = (const float*)d_in[pi++];
    const float* cf1_w   = (const float*)d_in[pi++];
    const float* cf1_b   = (const float*)d_in[pi++];
    const float* cf2_w   = (const float*)d_in[pi++];
    const float* cf2_b   = (const float*)d_in[pi++];
    const float* cf_g    = (const float*)d_in[pi++];
    const float* cf_bt   = (const float*)d_in[pi++];
    const float* convm_w = (const float*)d_in[pi++];
    const float* bnm_g   = (const float*)d_in[pi++];
    const float* bnm_bt  = (const float*)d_in[pi++];
    const float* convp_w = (const float*)d_in[pi++];
    const float* bnp_g   = (const float*)d_in[pi++];
    const float* bnp_bt  = (const float*)d_in[pi++];
    const float* em_W    = (const float*)d_in[pi++];
    const float* beta_u  = (const float*)d_in[pi++];
    const float* beta_a  = (const float*)d_in[pi++];
    const float* caps_g  = (const float*)d_in[pi++];
    const float* caps_bt = (const float*)d_in[pi++];
    const float* rec_w   = (const float*)d_in[pi++];
    const float* rec_g   = (const float*)d_in[pi++];
    const float* rec_bt  = (const float*)d_in[pi++];
    const float* c1_w    = (const float*)d_in[pi++];
    const float* bn1_g   = (const float*)d_in[pi++];
    const float* bn1_bt  = (const float*)d_in[pi++];
    const float* c2_w    = (const float*)d_in[pi++];
    const float* bn2_g   = (const float*)d_in[pi++];
    const float* bn2_bt  = (const float*)d_in[pi++];
    const float* c3_w    = (const float*)d_in[pi++];
    const float* bn3_g   = (const float*)d_in[pi++];
    const float* bn3_bt  = (const float*)d_in[pi++];
    const float* f1a_w   = (const float*)d_in[pi++];
    const float* f1a_b   = (const float*)d_in[pi++];
    const float* f1b_w   = (const float*)d_in[pi++];
    const float* f1b_b   = (const float*)d_in[pi++];
    const float* f1_g    = (const float*)d_in[pi++];
    const float* f1_bt   = (const float*)d_in[pi++];
    const float* f2a_w   = (const float*)d_in[pi++];
    const float* f2a_b   = (const float*)d_in[pi++];
    const float* f2b_w   = (const float*)d_in[pi++];
    const float* f2b_b   = (const float*)d_in[pi++];
    const float* f2_g    = (const float*)d_in[pi++];
    const float* f2_bt   = (const float*)d_in[pi++];
    const float* sa1_w   = (const float*)d_in[pi++];

    const int Bn = 4;
    float* W = (float*)d_ws;
    size_t off = 0;
    auto alloc = [&](size_t n) {
        off = (off + 15) & ~(size_t)15;   // 64B alignment for b128 loads
        float* p = W + off; off += n; return p;
    };

    // bf16 scratch (reused sequentially by every conv)
    __bf16* nhwcb = (__bf16*)alloc((size_t)Bn * 128 * 96 * 96 / 2 + 16); // 4.72M bf16
    __bf16* wgtb  = (__bf16*)alloc(110000);                              // 220K bf16

    // ---------------- fusion front-end ----------------
    float* xf = alloc((size_t)Bn * 192 * 48 * 48);
    resize_into<<<EWG(Bn * 64 * 48 * 48)>>>(in1, xf, Bn, 64, 24, 24, 48, 48, 192, 0);
    copy_into  <<<EWG(Bn * 64 * 48 * 48)>>>(in2, xf, Bn, 64, 48 * 48, 192, 64);
    resize_into<<<EWG(Bn * 64 * 48 * 48)>>>(in3, xf, Bn, 64, 96, 96, 48, 48, 192, 128);

    float* mp = alloc(Bn * 192);
    chan_spatial_max<<<Bn * 192, 256, 0, s>>>(xf, mp, 48 * 48);
    float* ca = alloc(Bn * 192);
    ca_kernel<<<Bn, 192, 0, s>>>(mp, ca_fc1, ca_fc2, ca);
    scale_ca<<<EWG(Bn * 192 * 48 * 48)>>>(xf, ca, 48 * 48, Bn * 192 * 48 * 48);

    float* t1 = alloc((size_t)Bn * 192 * 48 * 48);
    launch_conv(s, xf, cf1_w, cf1_b, t1, nhwcb, wgtb, Bn, 192, 48, 48, 192, 1, 1, 0);
    float* xf2 = alloc((size_t)Bn * 64 * 48 * 48);
    launch_conv(s, t1, cf2_w, cf2_b, xf2, nhwcb, wgtb, Bn, 192, 48, 48, 64, 3, 3, 1);

    float* mean = alloc(192);
    float* var  = alloc(192);
    run_bn(s, xf2, mean, var, cf_g, cf_bt, Bn, 64, 48 * 48, 1);

    // ---------------- capsule layer ----------------
    float* mbuf = alloc((size_t)Bn * 8 * 46 * 46);
    launch_conv(s, xf2, convm_w, nullptr, mbuf, nhwcb, wgtb, Bn, 64, 48, 48, 8, 5, 5, 1);
    run_bn(s, mbuf, mean, var, bnm_g, bnm_bt, Bn, 8, 46 * 46, 2);

    float* pbuf = alloc((size_t)Bn * 128 * 46 * 46);
    launch_conv(s, xf2, convp_w, nullptr, pbuf, nhwcb, wgtb, Bn, 64, 48, 48, 128, 5, 5, 1);
    run_bn(s, pbuf, mean, var, bnp_g, bnp_bt, Bn, 128, 46 * 46, 0);

    float* pose_o = alloc((size_t)Bn * 128 * 23 * 23);
    em_routing_kernel<<<Bn * 23 * 23, 128, 0, s>>>(mbuf, pbuf, em_W, beta_u, beta_a, pose_o);
    run_bn(s, pose_o, mean, var, caps_g, caps_bt, Bn, 128, 23 * 23, 0);

    float* rec = alloc((size_t)Bn * 64 * 23 * 23);
    launch_conv(s, pose_o, rec_w, nullptr, rec, nhwcb, wgtb, Bn, 128, 23, 23, 64, 3, 3, 1);
    run_bn(s, rec, mean, var, rec_g, rec_bt, Bn, 64, 23 * 23, 1);

    // ---------------- per-scale reconstruction ----------------
    float* cat1 = alloc((size_t)Bn * 128 * 24 * 24);
    copy_into  <<<EWG(Bn * 64 * 24 * 24)>>>(in1, cat1, Bn, 64, 24 * 24, 128, 0);
    resize_into<<<EWG(Bn * 64 * 24 * 24)>>>(rec, cat1, Bn, 64, 23, 23, 24, 24, 128, 64);
    float* o1 = alloc((size_t)Bn * 64 * 24 * 24);
    launch_conv(s, cat1, c1_w, nullptr, o1, nhwcb, wgtb, Bn, 128, 24, 24, 64, 3, 3, 1);
    run_bn(s, o1, mean, var, bn1_g, bn1_bt, Bn, 64, 24 * 24, 1);

    float* cat2 = alloc((size_t)Bn * 128 * 48 * 48);
    copy_into  <<<EWG(Bn * 64 * 48 * 48)>>>(in2, cat2, Bn, 64, 48 * 48, 128, 0);
    resize_into<<<EWG(Bn * 64 * 48 * 48)>>>(rec, cat2, Bn, 64, 23, 23, 48, 48, 128, 64);
    float* o2 = alloc((size_t)Bn * 64 * 48 * 48);
    launch_conv(s, cat2, c2_w, nullptr, o2, nhwcb, wgtb, Bn, 128, 48, 48, 64, 3, 3, 1);
    run_bn(s, o2, mean, var, bn2_g, bn2_bt, Bn, 64, 48 * 48, 1);

    float* cat3 = alloc((size_t)Bn * 128 * 96 * 96);
    copy_into  <<<EWG(Bn * 64 * 96 * 96)>>>(in3, cat3, Bn, 64, 96 * 96, 128, 0);
    resize_into<<<EWG(Bn * 64 * 96 * 96)>>>(rec, cat3, Bn, 64, 23, 23, 96, 96, 128, 64);
    float* o3 = alloc((size_t)Bn * 64 * 96 * 96);
    launch_conv(s, cat3, c3_w, nullptr, o3, nhwcb, wgtb, Bn, 128, 96, 96, 64, 3, 3, 1);
    run_bn(s, o3, mean, var, bn3_g, bn3_bt, Bn, 64, 96 * 96, 1);

    // ---------------- cascade fusion, stage 1 (48x48) ----------------
    const int n48 = Bn * 64 * 48 * 48;
    float* o1r = alloc((size_t)n48);
    resize_into<<<EWG(n48)>>>(o1, o1r, Bn, 64, 24, 24, 48, 48, 64, 0);
    float* mx48 = alloc(Bn * 48 * 48);
    float* sa48 = alloc(Bn * 48 * 48);
    max_over_c<<<EWG(Bn * 48 * 48)>>>(o1r, mx48, 64, 48 * 48, Bn * 48 * 48);
    conv7_sig <<<EWG(Bn * 48 * 48)>>>(mx48, sa1_w, sa48, 48, 48, Bn * 48 * 48);
    apply_sa  <<<EWG(n48)>>>(o1r, sa48, 64, 48 * 48, n48);

    float* prod = alloc((size_t)n48);
    ew_mul<<<EWG(n48)>>>(o2, o1r, prod, n48);
    float* fa = alloc((size_t)n48);
    launch_conv(s, prod, f1a_w, f1a_b, fa, nhwcb, wgtb, Bn, 64, 48, 48, 64, 1, 1, 0);
    float* fb = alloc((size_t)n48);
    launch_conv(s, fa, f1b_w, f1b_b, fb, nhwcb, wgtb, Bn, 64, 48, 48, 64, 3, 3, 1);
    run_bn(s, fb, mean, var, f1_g, f1_bt, Bn, 64, 48 * 48, 1);
    ew_add_into<<<EWG(n48)>>>(o2, fb, n48);

    // ---------------- cascade fusion, stage 2 (96x96) ----------------
    const int n96 = Bn * 64 * 96 * 96;
    float* o2r = alloc((size_t)n96);
    resize_into<<<EWG(n96)>>>(o2, o2r, Bn, 64, 48, 48, 96, 96, 64, 0);
    float* mx96 = alloc(Bn * 96 * 96);
    float* sa96 = alloc(Bn * 96 * 96);
    max_over_c<<<EWG(Bn * 96 * 96)>>>(o2r, mx96, 64, 96 * 96, Bn * 96 * 96);
    conv7_sig <<<EWG(Bn * 96 * 96)>>>(mx96, sa1_w, sa96, 96, 96, Bn * 96 * 96);
    apply_sa  <<<EWG(n96)>>>(o2r, sa96, 64, 96 * 96, n96);

    float* prod96 = alloc((size_t)n96);
    ew_mul<<<EWG(n96)>>>(o3, o2r, prod96, n96);
    float* ga = alloc((size_t)n96);
    launch_conv(s, prod96, f2a_w, f2a_b, ga, nhwcb, wgtb, Bn, 64, 96, 96, 64, 1, 1, 0);
    float* gb = alloc((size_t)n96);
    launch_conv(s, ga, f2b_w, f2b_b, gb, nhwcb, wgtb, Bn, 64, 96, 96, 64, 3, 3, 1);
    run_bn(s, gb, mean, var, f2_g, f2_bt, Bn, 64, 96 * 96, 1);
    ew_add_into<<<EWG(n96)>>>(o3, gb, n96);

    // ---------------- outputs: (out1@48, out2@96, out3@96) ----------------
    float* out = (float*)d_out;
    hipMemcpyAsync(out,             o1r, (size_t)n48 * 4, hipMemcpyDeviceToDevice, s);
    hipMemcpyAsync(out + n48,       o2r, (size_t)n96 * 4, hipMemcpyDeviceToDevice, s);
    hipMemcpyAsync(out + n48 + n96, o3,  (size_t)n96 * 4, hipMemcpyDeviceToDevice, s);
}
